// MOSNet_59485297049779
// MI455X (gfx1250) — compile-verified, matrix operand order checked by reading, not run on hardware
//
#include <hip/hip_runtime.h>
#include <hip/hip_bf16.h>
#include <math.h>

typedef __bf16 bf16;
typedef __attribute__((ext_vector_type(16))) __bf16 v16bf;
typedef __attribute__((ext_vector_type(8)))  float  v8f;

// ---------- helpers ----------
__device__ __forceinline__ bf16 tobf(float f) {
    union { float f; unsigned u; } v; v.f = f;
    unsigned r = v.u + 0x7fffu + ((v.u >> 16) & 1u);   // round-to-nearest-even
    union { unsigned short s; bf16 b; } w; w.s = (unsigned short)(r >> 16);
    return w.b;
}
__device__ __forceinline__ float sigm(float x) { return 1.0f / (1.0f + expf(-x)); }

// Load a 16x32 bf16 tile (row-major, leading dim ld) into WMMA A/B lane layout.
// lanes 0-15: row = lane, K in {0..7,16..23}; lanes 16-31: row = lane-16, K in {8..15,24..31}
__device__ __forceinline__ v16bf load_tile(const bf16* __restrict__ p, int ld, int lane) {
    v16bf v;
    const int r  = lane & 15;
    const int kb = (lane & 16) ? 8 : 0;
#pragma unroll
    for (int i = 0; i < 16; ++i) {
        int k = ((i < 8) ? i : (i + 8)) + kb;
        v[i] = p[r * ld + k];
    }
    return v;
}

#define WMMA_BF16(a, b, c) \
    __builtin_amdgcn_wmma_f32_16x16x32_bf16(false, (a), false, (b), (short)0, (c), false, false)

// ---------- f32 -> bf16 convert ----------
__global__ void cvt_bf16(const float* __restrict__ s, bf16* __restrict__ d, long n) {
    long i = (long)blockIdx.x * blockDim.x + threadIdx.x;
    if (i < n) d[i] = tobf(s[i]);
}

// f32 [COUT][K] -> bf16 [COUT][KPAD] with zero padding (K -> next multiple of 32)
__global__ void cvt_pad(const float* __restrict__ s, bf16* __restrict__ d,
                        int K, int KPAD, int n) {
    int i = blockIdx.x * 256 + threadIdx.x;
    if (i >= n) return;
    int co = i / KPAD, k = i % KPAD;
    d[i] = (k < K) ? tobf(s[co * K + k]) : tobf(0.0f);
}

// ---------- implicit-GEMM conv (3x3, stride (1,3), pad (1,1), bias+ReLU) ----------
// in : [S=256][CIN][T=256][WIN] bf16, w: [COUT][KPAD] bf16 (zero-padded),
// out bf16:  FEAT=0: [S][COUT][T][WOUT]   FEAT=1 (WOUT==1): [S][T][COUT]
template <int CIN, int COUT, int WIN, int WOUT, bool FEAT>
__global__ __launch_bounds__(256) void conv_wmma(const bf16* __restrict__ in,
                                                 const bf16* __restrict__ w,
                                                 const float* __restrict__ bias,
                                                 bf16* __restrict__ out) {
    const int  K      = CIN * 9;
    const int  KC     = (K + 31) / 32;
    const int  KPAD   = KC * 32;
    const int  NT     = COUT / 16;
    const long ROWS   = 256L * 256L * WOUT;
    const long NTILES = (ROWS / 16) * NT;

    const int lane = threadIdx.x & 31;
    const int wv   = threadIdx.x >> 5;
    long tile = (long)blockIdx.x * 8 + wv;
    if (tile >= NTILES) return;                 // wave-uniform exit, EXEC stays all-ones
    const int  tN = (int)(tile % NT);
    const long tM = tile / NT;

    // per-lane A row decomposition
    const long rA   = tM * 16 + (lane & 15);
    const int  sA   = (int)(rA / (256L * WOUT));
    const int  remA = (int)(rA % (256L * WOUT));
    const int  tA   = remA / WOUT;
    const int  woA  = remA % WOUT;
    const int  kb   = (lane & 16) ? 8 : 0;
    const bf16* wB  = w + (long)(tN * 16 + (lane & 15)) * KPAD;

    v8f acc = {};
#pragma unroll
    for (int kc = 0; kc < KC; ++kc) {
        const int k0 = kc * 32;
        v16bf a, b;
#pragma unroll
        for (int i = 0; i < 16; ++i) {
            int k = ((i < 8) ? i : (i + 8)) + kb + k0;
            // A: im2col gather with zero padding
            bf16 av = tobf(0.0f);
            if (k < K) {
                int ci = k / 9, kr = k % 9;
                int kh = kr / 3, kw = kr % 3;
                int ti = tA + kh - 1;
                int mi = woA * 3 - 1 + kw;
                if (ti >= 0 && ti < 256 && mi >= 0 && mi < WIN)
                    av = in[(((long)sA * CIN + ci) * 256 + ti) * WIN + mi];
            }
            a[i] = av;
            b[i] = wB[k];                       // zero-padded, no mask needed
        }
        acc = WMMA_BF16(a, b, acc);
    }

    // store D with bias + ReLU
    const int n  = lane & 15;
    const int co = tN * 16 + n;
    const int mb = (lane >> 4) * 8;
    const float bco = bias[co];
#pragma unroll
    for (int i = 0; i < 8; ++i) {
        long r   = tM * 16 + mb + i;
        int  s   = (int)(r / (256L * WOUT));
        int  rem = (int)(r % (256L * WOUT));
        int  t   = rem / WOUT;
        int  wo  = rem % WOUT;
        float v = acc[i] + bco;
        v = v > 0.0f ? v : 0.0f;
        long oidx = FEAT ? (((long)s * 256 + t) * COUT + co)
                         : ((((long)s * COUT + co) * 256 + t) * WOUT + wo);
        out[oidx] = tobf(v);
    }
}

// ---------- input projection GEMM: z = feat @ Wih^T + (bih + bhh) ----------
// feat [65536 x 128] bf16 (row = s*256+t), Wih [512 x 128] bf16, z [65536 x 512] f32
__global__ __launch_bounds__(256) void xproj_wmma(const bf16* __restrict__ feat,
                                                  const bf16* __restrict__ Wih,
                                                  const float* __restrict__ bih,
                                                  const float* __restrict__ bhh,
                                                  float* __restrict__ z) {
    const int lane = threadIdx.x & 31;
    const int wv   = threadIdx.x >> 5;
    long tile = (long)blockIdx.x * 8 + wv;           // 4096 Mtiles * 32 Ntiles
    if (tile >= 4096L * 32) return;
    const int  tN = (int)(tile & 31);
    const long tM = tile >> 5;

    const int n  = lane & 15;
    const int j  = tN * 16 + n;
    const float bj = bih[j] + bhh[j];
    v8f acc;
#pragma unroll
    for (int i = 0; i < 8; ++i) acc[i] = bj;

    const bf16* A = feat + tM * 16 * 128;
    const bf16* B = Wih + (long)tN * 16 * 128;
#pragma unroll
    for (int kc = 0; kc < 4; ++kc) {
        v16bf a = load_tile(A + kc * 32, 128, lane);
        v16bf b = load_tile(B + kc * 32, 128, lane);
        acc = WMMA_BF16(a, b, acc);
    }
    const int mb = (lane >> 4) * 8;
#pragma unroll
    for (int i = 0; i < 8; ++i)
        z[(tM * 16 + mb + i) * 512 + j] = acc[i];
}

// ---------- LSTM scan (16 sequences per workgroup, WMMA recurrent matmul) ----------
// zin [row = s*256+t][512] f32 (x-proj + bias), Whh [512 x 128] bf16 (LDS-resident),
// hbuf [s][t][256] bf16, dirofs 0 (fwd) / 128 (bwd)
// dynamic LDS: Whh 131072B | zsh 32768B | csh 8192B | hsh 4096B  = 176128B (< 320KB WGP LDS)
#define LSTM_LDS_BYTES (131072 + 32768 + 8192 + 4096)
template <bool REV>
__global__ __launch_bounds__(256) void lstm_wmma(const float* __restrict__ zin,
                                                 const bf16* __restrict__ Whh,
                                                 bf16* __restrict__ hbuf,
                                                 int dirofs) {
    extern __shared__ char smem[];
    bf16*  whs = (bf16*)smem;                          // [512 x 128]
    float* zsh = (float*)(smem + 131072);              // [16 x 512]
    float* csh = (float*)(smem + 131072 + 32768);      // [16 x 128]
    bf16*  hsh = (bf16*)(smem + 131072 + 32768 + 8192);// [16 x 128]

    const int tid  = threadIdx.x;
    const int lane = tid & 31;
    const int wv   = tid >> 5;
    const int s0   = blockIdx.x * 16;

    // preload Whh into LDS (8192 x b128) + init h,c
    for (int i = tid; i < 8192; i += 256) ((int4*)whs)[i] = ((const int4*)Whh)[i];
    for (int i = tid; i < 16 * 128; i += 256) { hsh[i] = tobf(0.0f); csh[i] = 0.0f; }
    __syncthreads();

    for (int step = 0; step < 256; ++step) {
        const int t = REV ? (255 - step) : step;

        // prefetch next step's x-projection rows while we compute this step
        const int tn = REV ? (t - 1) : (t + 1);
        if (tn >= 0 && tn < 256) {
            const float* pz = &zin[((long)(s0 + (lane & 15)) * 256 + tn) * 512 + wv * 64];
            __builtin_prefetch(pz, 0, 0);
            __builtin_prefetch(pz + 32, 0, 0);
        }

        // z = h @ Whh^T (+ precomputed x-proj as accumulator init)
        v16bf a[4];
#pragma unroll
        for (int kc = 0; kc < 4; ++kc) a[kc] = load_tile(hsh + kc * 32, 128, lane);

        const int n  = lane & 15;
        const int mb = (lane >> 4) * 8;
#pragma unroll
        for (int jt = 0; jt < 4; ++jt) {
            const int n0 = wv * 64 + jt * 16;
            v8f acc;
#pragma unroll
            for (int i = 0; i < 8; ++i)
                acc[i] = zin[((long)(s0 + mb + i) * 256 + t) * 512 + n0 + n];
#pragma unroll
            for (int kc = 0; kc < 4; ++kc) {
                v16bf b = load_tile(whs + n0 * 128 + kc * 32, 128, lane);
                acc = WMMA_BF16(a[kc], b, acc);
            }
#pragma unroll
            for (int i = 0; i < 8; ++i) zsh[(mb + i) * 512 + n0 + n] = acc[i];
        }
        __syncthreads();

        // gate nonlinearity: 2048 cells / 256 threads
#pragma unroll
        for (int e = 0; e < 8; ++e) {
            const int cell = e * 256 + tid;
            const int m = cell >> 7, j = cell & 127;
            const float zi = zsh[m * 512 + j];
            const float zf = zsh[m * 512 + 128 + j];
            const float zg = zsh[m * 512 + 256 + j];
            const float zo = zsh[m * 512 + 384 + j];
            float c = sigm(zf) * csh[m * 128 + j] + sigm(zi) * tanhf(zg);
            float h = sigm(zo) * tanhf(c);
            csh[m * 128 + j] = c;
            bf16 hb = tobf(h);
            hsh[m * 128 + j] = hb;
            hbuf[((long)(s0 + m) * 256 + t) * 256 + dirofs + j] = hb;
        }
        __syncthreads();
    }
}

// ---------- FC1 split-K: [256 x 65536] @ [65536 x 128], 16 K-segments ----------
// partials: pbuf [16][256][128] f32
__global__ __launch_bounds__(256) void fc1_wmma(const bf16* __restrict__ hflat,
                                                const bf16* __restrict__ w,   // [128 x 65536]
                                                float* __restrict__ pbuf) {
    const int lane = threadIdx.x & 31;
    const int wv   = threadIdx.x >> 5;
    const int tile = blockIdx.x * 8 + wv;   // [seg 16][tM 16][tN 8] = 2048 waves
    if (tile >= 2048) return;
    const int tN = tile & 7, tM = (tile >> 3) & 15, seg = tile >> 7;

    v8f acc = {};
    const bf16* A = hflat + (long)tM * 16 * 65536 + seg * 4096;
    const bf16* B = w + (long)tN * 16 * 65536 + seg * 4096;
    const int r = lane & 15;
    for (int kc = 0; kc < 128; ++kc) {
        __builtin_prefetch(A + (long)r * 65536 + kc * 32 + 1024, 0, 0);
        __builtin_prefetch(B + (long)r * 65536 + kc * 32 + 1024, 0, 0);
        v16bf a  = load_tile(A + kc * 32, 65536, lane);
        v16bf bb = load_tile(B + kc * 32, 65536, lane);
        acc = WMMA_BF16(a, bb, acc);
    }
    const int n  = lane & 15;
    const int mb = (lane >> 4) * 8;
#pragma unroll
    for (int i = 0; i < 8; ++i)
        pbuf[seg * 32768 + (tM * 16 + mb + i) * 128 + tN * 16 + n] = acc[i];
}

// reduce 16 partials + bias + ReLU -> zfc [256 x 128]
__global__ void fc1_reduce(const float* __restrict__ pbuf, const float* __restrict__ b,
                           float* __restrict__ zout) {
    int i = blockIdx.x * 256 + threadIdx.x;
    if (i >= 32768) return;
    float s = 0.0f;
#pragma unroll
    for (int g = 0; g < 16; ++g) s += pbuf[g * 32768 + i];
    s += b[i & 127];
    zout[i] = s > 0.0f ? s : 0.0f;
}

// ---------- scoring head + per-utterance mean ----------
__global__ void finalize_k(const float* __restrict__ z, const float* __restrict__ fsw,
                           const float* __restrict__ fsb, float* __restrict__ out) {
    __shared__ float sc[256];
    const int tid = threadIdx.x;     // == sequence index s = b*16 + f
    float acc = 0.0f;
    for (int j = 0; j < 128; ++j) acc += z[tid * 128 + j] * fsw[j];
    acc += fsb[0];
    float score = 4.0f * sigm(acc) + 1.0f;
    sc[tid] = score;
    out[16 + tid] = score;           // frames [16][16]
    __syncthreads();
    if (tid < 16) {
        float m = 0.0f;
        for (int f = 0; f < 16; ++f) m += sc[tid * 16 + f];
        out[tid] = m * (1.0f / 16.0f);   // utt
    }
}

// ---------- host-side orchestration ----------
extern "C" void kernel_launch(void* const* d_in, const int* in_sizes, int n_in,
                              void* d_out, int out_size, void* d_ws, size_t ws_size,
                              hipStream_t stream) {
    const float* audio = (const float*)d_in[0];
    const float* w0 = (const float*)d_in[2];  const float* b0 = (const float*)d_in[3];
    const float* w1 = (const float*)d_in[4];  const float* b1 = (const float*)d_in[5];
    const float* w2 = (const float*)d_in[6];  const float* b2 = (const float*)d_in[7];
    const float* w3 = (const float*)d_in[8];  const float* b3 = (const float*)d_in[9];
    const float* Wih_f = (const float*)d_in[10]; const float* Whh_f = (const float*)d_in[11];
    const float* bih_f = (const float*)d_in[12]; const float* bhh_f = (const float*)d_in[13];
    const float* Wih_b = (const float*)d_in[14]; const float* Whh_b = (const float*)d_in[15];
    const float* bih_b = (const float*)d_in[16]; const float* bhh_b = (const float*)d_in[17];
    const float* fc1_w = (const float*)d_in[18]; const float* fc1_b = (const float*)d_in[19];
    const float* fs_w  = (const float*)d_in[20]; const float* fs_b  = (const float*)d_in[21];
    float* out = (float*)d_out;

    // workspace carve-out (256B aligned)
    char* ws = (char*)d_ws;
    size_t off = 0;
    auto carve = [&](size_t bytes) { char* p = ws + off; off = (off + bytes + 255) & ~(size_t)255; return p; };
    bf16* abf   = (bf16*)carve(5308416L * 2);        // audio bf16 [256][256][81]
    bf16* w0p   = (bf16*)carve(512L * 2);            // 16 x 32   (K=9   padded)
    bf16* w1p   = (bf16*)carve(5120L * 2);           // 32 x 160  (K=144 padded)
    bf16* w2p   = (bf16*)carve(18432L * 2);          // 64 x 288
    bf16* w3p   = (bf16*)carve(73728L * 2);          // 128 x 576
    bf16* wihfb = (bf16*)carve(65536L * 2);
    bf16* whhfb = (bf16*)carve(65536L * 2);
    bf16* wihbb = (bf16*)carve(65536L * 2);
    bf16* whhbb = (bf16*)carve(65536L * 2);
    bf16* fc1b  = (bf16*)carve(8388608L * 2);
    bf16* act0  = (bf16*)carve(28311552L * 2);       // [256][16][256][27]
    bf16* act1  = (bf16*)carve(18874368L * 2);       // [256][32][256][9]
    bf16* act2  = (bf16*)carve(12582912L * 2);       // [256][64][256][3]
    bf16* feat  = (bf16*)carve(8388608L * 2);        // [s*256+t][128]
    float* zbuf = (float*)carve(33554432L * 4);      // [65536][512], reused fwd/bwd
    bf16* hbuf  = (bf16*)carve(16777216L * 2);       // [s][t][256]
    float* pbuf = (float*)carve(524288L * 4);        // [16][256][128] fc1 partials
    float* zfc  = (float*)carve(32768L * 4);         // [256][128]

    auto cvt = [&](const float* s, bf16* d, long n) {
        cvt_bf16<<<(int)((n + 255) / 256), 256, 0, stream>>>(s, d, n);
    };
    auto cvtp = [&](const float* s, bf16* d, int K, int KPAD, int COUT) {
        int n = COUT * KPAD;
        cvt_pad<<<(n + 255) / 256, 256, 0, stream>>>(s, d, K, KPAD, n);
    };
    cvt(audio, abf, 5308416L);
    cvtp(w0, w0p, 9, 32, 16);
    cvtp(w1, w1p, 144, 160, 32);
    cvtp(w2, w2p, 288, 288, 64);
    cvtp(w3, w3p, 576, 576, 128);
    cvt(Wih_f, wihfb, 65536); cvt(Whh_f, whhfb, 65536);
    cvt(Wih_b, wihbb, 65536); cvt(Whh_b, whhbb, 65536);
    cvt(fc1_w, fc1b, 8388608L);

    auto nblk = [](long tiles) { return (int)((tiles + 7) / 8); };
    conv_wmma<1, 16, 81, 27, false><<<nblk(110592L * 1), 256, 0, stream>>>(abf,  w0p, b0, act0);
    conv_wmma<16, 32, 27, 9, false><<<nblk(36864L * 2),  256, 0, stream>>>(act0, w1p, b1, act1);
    conv_wmma<32, 64, 9, 3, false><<<nblk(12288L * 4),   256, 0, stream>>>(act1, w2p, b2, act2);
    conv_wmma<64, 128, 3, 1, true><<<nblk(4096L * 8),    256, 0, stream>>>(act2, w3p, b3, feat);

    xproj_wmma<<<nblk(4096L * 32), 256, 0, stream>>>(feat, wihfb, bih_f, bhh_f, zbuf);
    lstm_wmma<false><<<16, 256, LSTM_LDS_BYTES, stream>>>(zbuf, whhfb, hbuf, 0);
    xproj_wmma<<<nblk(4096L * 32), 256, 0, stream>>>(feat, wihbb, bih_b, bhh_b, zbuf);
    lstm_wmma<true><<<16, 256, LSTM_LDS_BYTES, stream>>>(zbuf, whhbb, hbuf, 128);

    fc1_wmma<<<256, 256, 0, stream>>>(hbuf, fc1b, pbuf);
    fc1_reduce<<<128, 256, 0, stream>>>(pbuf, fc1_b, zfc);
    finalize_k<<<1, 256, 0, stream>>>(zfc, fs_w, fs_b, out);
}
